// LIFfeatureLayer_42030549959279
// MI455X (gfx1250) — matrix operand order repaired and checked by reading, not compiled, not run on hardware
//
#include <hip/hip_runtime.h>

typedef _Float16 v16h __attribute__((ext_vector_type(16)));
typedef _Float16 v8h  __attribute__((ext_vector_type(8)));
typedef float    v8f  __attribute__((ext_vector_type(8)));

#define BB 32
#define TT 2048
#define II 256
#define HH 512
#define LDSS 264            // 256 f16 + 16B pad per row -> 528B stride
#define NTILE (TT / 16)     // 128 timestep tiles

#define ALPHA_MIN_ 0.8187307530779818f
#define ALPHA_MAX_ 0.9607894391523232f
#define SWAPX16 0x401f      // ds_swizzle: xor_mask=0x10, and_mask=0x1f

__device__ __forceinline__ float swap16(float v) {
  return __int_as_float(__builtin_amdgcn_ds_swizzle(__float_as_int(v), SWAPX16));
}

// convert 16 consecutive f32 -> v16h
__device__ __forceinline__ v16h cvt16(const float* p) {
  const float4* q = (const float4*)p;
  float4 q0 = q[0], q1 = q[1], q2 = q[2], q3 = q[3];
  v16h f;
  f[0]=(_Float16)q0.x;  f[1]=(_Float16)q0.y;  f[2]=(_Float16)q0.z;  f[3]=(_Float16)q0.w;
  f[4]=(_Float16)q1.x;  f[5]=(_Float16)q1.y;  f[6]=(_Float16)q1.z;  f[7]=(_Float16)q1.w;
  f[8]=(_Float16)q2.x;  f[9]=(_Float16)q2.y;  f[10]=(_Float16)q2.z; f[11]=(_Float16)q2.w;
  f[12]=(_Float16)q3.x; f[13]=(_Float16)q3.y; f[14]=(_Float16)q3.z; f[15]=(_Float16)q3.w;
  return f;
}

__global__ __launch_bounds__(64)
void lif_fused_kernel(const float* __restrict__ x,
                      const float* __restrict__ W,
                      const float* __restrict__ alpha,
                      const float* __restrict__ u0,
                      const float* __restrict__ s0,
                      float* __restrict__ out)
{
  __shared__ _Float16 xs[2][16 * LDSS];   // double-buffered f16 x-tile

  const int b    = blockIdx.y;
  const int tid  = threadIdx.x;
  const int wv   = tid >> 5;
  const int lane = tid & 31;
  const int col  = lane & 15;
  const int half = lane >> 4;
  const int hw   = blockIdx.x * 64 + wv * 32;   // wave's first h column

  // ---- B fragments: W rows for 2 column-tiles x 8 K-chunks, f32 -> f16 ----
  v16h bfrag[2][8];
  #pragma unroll
  for (int j = 0; j < 2; ++j) {
    const float* wrow = W + (size_t)(hw + j * 16 + col) * II;
    #pragma unroll
    for (int k = 0; k < 8; ++k)
      bfrag[j][k] = cvt16(wrow + k * 32 + half * 16);
  }

  // ---- per-lane LIF state: lane owns column h = hw + lane ----
  // carried state d = u - s  (absorbs the non-binary initial s0)
  const int hcol = hw + lane;
  float a_ = alpha[hcol];
  a_ = fminf(fmaxf(a_, ALPHA_MIN_), ALPHA_MAX_);
  const float bb = 1.0f - a_;
  float d = u0[b * HH + hcol] - s0[b * HH + hcol];

  const float* xb     = x + (size_t)b * TT * II;
  float*       outcol = out + (size_t)b * TT * HH + hcol;

  const int srow = tid >> 2;          // staging: 4 threads/row, 64 f32 each
  const int skof = (tid & 3) * 64;

  // stage x[b, t0:t0+16, 0:256] f32 -> f16 into LDS buffer
  auto stage = [&](int t0, _Float16* buf) {
    const float* src = xb + (size_t)(t0 + srow) * II + skof;
    _Float16*    dst = buf + srow * LDSS + skof;
    #pragma unroll
    for (int i = 0; i < 8; ++i) {
      float4 v0 = ((const float4*)src)[2 * i];
      float4 v1 = ((const float4*)src)[2 * i + 1];
      v8h h;
      h[0]=(_Float16)v0.x; h[1]=(_Float16)v0.y; h[2]=(_Float16)v0.z; h[3]=(_Float16)v0.w;
      h[4]=(_Float16)v1.x; h[5]=(_Float16)v1.y; h[6]=(_Float16)v1.z; h[7]=(_Float16)v1.w;
      *(v8h*)(dst + i * 8) = h;
    }
  };

  // 16x16x256 GEMM tile: two accumulators over 8 k32 chunks
  auto compute = [&](const _Float16* buf, v8f& c0, v8f& c1) {
    c0 = (v8f){0.f,0.f,0.f,0.f,0.f,0.f,0.f,0.f};
    c1 = (v8f){0.f,0.f,0.f,0.f,0.f,0.f,0.f,0.f};
    const _Float16* arow = buf + col * LDSS + half * 16;
    #pragma unroll
    for (int k = 0; k < 8; ++k) {
      v8h lo8 = *(const v8h*)(arow + k * 32);
      v8h hi8 = *(const v8h*)(arow + k * 32 + 8);
      v16h a = __builtin_shufflevector(lo8, hi8, 0,1,2,3,4,5,6,7,8,9,10,11,12,13,14,15);
      c0 = __builtin_amdgcn_wmma_f32_16x16x32_f16(
             false, a, false, bfrag[0][k], (short)0, c0, false, false);
      c1 = __builtin_amdgcn_wmma_f32_16x16x32_f16(
             false, a, false, bfrag[1][k], (short)0, c1, false, false);
    }
  };

  // LIF scan of one 16-step tile; lane scans its own column once.
  // C layout: lanes 0-15 hold t=r, lanes 16-31 hold t=8+r (per vgpr r).
  // Each lane swizzles only the register its partner needs (8 DS ops):
  //   half=0 receives partner's A0 (t=8..15 of tile0 col)
  //   half=1 receives partner's A1 (t=0..7  of tile1 col)
  auto scan_store = [&](v8f A0, v8f A1, int tbase) {
    float w[8];
    #pragma unroll
    for (int r = 0; r < 8; ++r) {
      float m = half ? A0[r] : A1[r];      // what my partner needs
      w[r] = swap16(m);                     // what I receive
    }
    float bw[16];                           // off-chain b * wx products
    #pragma unroll
    for (int r = 0; r < 8; ++r) {
      bw[r]     = bb * (half ? w[r]  : A0[r]);
      bw[8 + r] = bb * (half ? A1[r] : w[r]);
    }
    float* op = outcol + (size_t)tbase * HH;
    #pragma unroll
    for (int t = 0; t < 16; ++t) {
      float u  = fmaf(a_, d, bw[t]);        // u_t = a*(u-s) + b*wx
      bool  sb = u > 1.0f;                  // heaviside spike
      d = sb ? (u - 1.0f) : u;              // d_t = u_t - s_t (3-op chain)
      __builtin_nontemporal_store(sb ? 1.0f : 0.0f, op + (size_t)t * HH);
    }
  };

  // ---- software-pipelined main loop: WMMA(i) issued before scan(i-1) ----
  v8f p0, p1, c0, c1;
  stage(0, xs[0]);
  __syncthreads();
  compute(xs[0], p0, p1);

  for (int i = 1; i < NTILE; ++i) {
    stage(i * 16, xs[i & 1]);
    __syncthreads();
    compute(xs[i & 1], c0, c1);            // XDL work issued first...
    scan_store(p0, p1, (i - 1) * 16);      // ...scan VALU co-executes
    p0 = c0; p1 = c1;
  }
  scan_store(p0, p1, TT - 16);
}

extern "C" void kernel_launch(void* const* d_in, const int* in_sizes, int n_in,
                              void* d_out, int out_size, void* d_ws, size_t ws_size,
                              hipStream_t stream) {
  const float* x     = (const float*)d_in[0];
  const float* W     = (const float*)d_in[1];
  const float* alpha = (const float*)d_in[2];
  const float* u0    = (const float*)d_in[3];
  const float* s0    = (const float*)d_in[4];
  float* out = (float*)d_out;

  dim3 grid(HH / 64, BB);   // 8 h-tiles x 32 batches = 256 blocks
  lif_fused_kernel<<<grid, 64, 0, stream>>>(x, W, alpha, u0, s0, out);
}
